// RNNLayer_76089640616410
// MI455X (gfx1250) — matrix operand order repaired and checked by reading, not compiled, not run on hardware
//
#include <hip/hip_runtime.h>

typedef __bf16 bf16_t;
typedef __attribute__((ext_vector_type(16))) __bf16 v16bf;
typedef __attribute__((ext_vector_type(8)))  __bf16 v8bf;
typedef __attribute__((ext_vector_type(8)))  float  v8f;
typedef __attribute__((ext_vector_type(4)))  float  v4f;

constexpr int B = 32, T = 2048, I = 256, H = 256;

// ---------------------------------------------------------------------------
// Kernel 0: convert W_ih [H][I] and W_hh [H][H] f32 -> bf16 (row-major kept,
// since this code's B operand reads 16 contiguous K elements of a row of W).
// ---------------------------------------------------------------------------
__global__ __launch_bounds__(256)
void cvt_weights_kernel(const float* __restrict__ Wih, const float* __restrict__ Whh,
                        bf16_t* __restrict__ wsWih, bf16_t* __restrict__ wsWhh) {
    int i = blockIdx.x * blockDim.x + threadIdx.x;
    if (i < H * I) wsWih[i] = (bf16_t)Wih[i];
    if (i < H * H) wsWhh[i] = (bf16_t)Whh[i];
}

// ---------------------------------------------------------------------------
// Kernel 1: xproj[m][h] = sum_i x[m][i] * W_ih[h][i] + b_ih[h], m = b*T + t.
// One block = one 16-row M-tile staged once into LDS as bf16 (inputs read
// exactly once from HBM). 8 waves x 2 N-tiles cover all 256 columns.
// Result written straight into d_out (outs region) for in-place phase 2.
// ---------------------------------------------------------------------------
__global__ __launch_bounds__(256)
void xproj_kernel(const float* __restrict__ x,     // [B*T][I] f32
                  const bf16_t* __restrict__ Wih,  // [H][I]  bf16 (L2-resident)
                  const float* __restrict__ bih,   // [H]
                  float* __restrict__ out)         // [B*T][H] f32
{
    __shared__ bf16_t aLds[16][I];                 // 8 KB
    const int lane = threadIdx.x & 31;
    const int half = lane >> 4;                    // which 16-lane group
    const int l16  = lane & 15;
    const int wave = threadIdx.x >> 5;             // 0..7
    const long mBase = (long)blockIdx.x * 16;      // grid.x = B*T/16

    // Stage A tile: 16 rows x 256 f32 -> bf16 LDS. 4096 elems / 256 thr = 16 ea.
    {
        int idx = threadIdx.x * 16;                // element index in tile
        int m = idx / I, k = idx % I;
        const float* src = x + (mBase + m) * I + k;
        v4f f0 = *(const v4f*)(src + 0);
        v4f f1 = *(const v4f*)(src + 4);
        v4f f2 = *(const v4f*)(src + 8);
        v4f f3 = *(const v4f*)(src + 12);
        v8bf p0, p1;
#pragma unroll
        for (int j = 0; j < 4; ++j) {
            p0[j] = (bf16_t)f0[j]; p0[4 + j] = (bf16_t)f1[j];
            p1[j] = (bf16_t)f2[j]; p1[4 + j] = (bf16_t)f3[j];
        }
        *(v8bf*)(&aLds[m][k])     = p0;
        *(v8bf*)(&aLds[m][k + 8]) = p1;
    }
    __syncthreads();

    const int n0 = (2 * wave)     * 16;            // first N-tile column base
    const int n1 = (2 * wave + 1) * 16;            // second N-tile column base
    v8f acc0 = {}, acc1 = {};

#pragma unroll
    for (int kk = 0; kk < I / 32; ++kk) {
        const int kb = kk * 32;
        // A operand: lane l%16 = row, K groups {kb+16*half..+7} and {kb+16*half+8..+15}
        const int g0 = kb + 16 * half;
        const int g1 = kb + 16 * half + 8;
        v8bf a0 = *(const v8bf*)(&aLds[l16][g0]);
        v8bf a1 = *(const v8bf*)(&aLds[l16][g1]);
        v16bf a;
#pragma unroll
        for (int j = 0; j < 8; ++j) { a[j] = a0[j]; a[8 + j] = a1[j]; }
        // B operand: lane reads 16 contiguous K elems of row (ntile*16 + l%16)
        v16bf b0 = *(const v16bf*)(Wih + (long)(n0 + l16) * I + kb + 16 * half);
        v16bf b1 = *(const v16bf*)(Wih + (long)(n1 + l16) * I + kb + 16 * half);
        acc0 = __builtin_amdgcn_wmma_f32_16x16x32_bf16(false, a, false, b0, (short)0, acc0, false, false);
        acc1 = __builtin_amdgcn_wmma_f32_16x16x32_bf16(false, a, false, b1, (short)0, acc1, false, false);
    }

    // C/D layout: VGPR r -> row (r + 8*half), col l%16 within tile.
#pragma unroll
    for (int r = 0; r < 8; ++r) {
        long m = mBase + r + 8 * half;
        out[m * H + n0 + l16] = acc0[r] + bih[n0 + l16];
        out[m * H + n1 + l16] = acc1[r] + bih[n1 + l16];
    }
}

// ---------------------------------------------------------------------------
// Kernel 2: persistent sequential scan. grid = 2 blocks x 512 threads;
// block g owns batch rows [16g, 16g+16). Wave w (of 16) owns N-tile w and
// keeps its W_hh slice (16 rows x 256 K, bf16) in 64 VGPRs for all T steps.
// Hidden state double-buffered in LDS (2 x 16 x 256 bf16 = 16 KB).
// xproj loads are software-pipelined one timestep ahead (L2-resident).
// ---------------------------------------------------------------------------
__global__ __launch_bounds__(512)
void scan_kernel(const bf16_t* __restrict__ Whh,  // [H][H] bf16
                 const float* __restrict__ h0,    // [B][H] f32
                 const float* __restrict__ bhh,   // [H]
                 float* __restrict__ out,         // [B][T][H], holds xproj on entry
                 float* __restrict__ hfin)        // [B][H]
{
    __shared__ bf16_t hA[2][16][H];                // 16 KB
    const int lane = threadIdx.x & 31;
    const int half = lane >> 4;
    const int l16  = lane & 15;
    const int wave = threadIdx.x >> 5;             // 0..15 -> N-tile
    const int bBase = blockIdx.x * 16;             // grid.x = 2
    const int ncol  = wave * 16 + l16;             // this lane's output column

    // Preload B operands for all 8 K-steps: W_hh row ncol, contiguous K.
    v16bf breg[8];
#pragma unroll
    for (int kk = 0; kk < 8; ++kk)
        breg[kk] = *(const v16bf*)(Whh + (long)ncol * H + kk * 32 + 16 * half);
    const float bias = bhh[ncol];

    // Initialize hA[0] from hidden_state (f32 -> bf16). 4096 elems / 512 thr.
    for (int idx = threadIdx.x; idx < 16 * H; idx += blockDim.x) {
        int m = idx / H, k = idx % H;
        hA[0][m][k] = (bf16_t)h0[(long)(bBase + m) * H + k];
    }
    __syncthreads();

    // Prefetch xproj for t=0. Lane/VGPR r -> batch row bBase + r + 8*half.
    float xp[8];
#pragma unroll
    for (int r = 0; r < 8; ++r)
        xp[r] = out[((long)(bBase + r + 8 * half) * T + 0) * H + ncol];

    int cur = 0;
    for (int t = 0; t < T; ++t) {
        v8f acc = {};
#pragma unroll
        for (int kk = 0; kk < 8; ++kk) {
            const int kb = kk * 32;
            const int g0 = kb + 16 * half;
            const int g1 = kb + 16 * half + 8;
            v8bf a0 = *(const v8bf*)(&hA[cur][l16][g0]);
            v8bf a1 = *(const v8bf*)(&hA[cur][l16][g1]);
            v16bf a;
#pragma unroll
            for (int j = 0; j < 8; ++j) { a[j] = a0[j]; a[8 + j] = a1[j]; }
            acc = __builtin_amdgcn_wmma_f32_16x16x32_bf16(false, a, false, breg[kk], (short)0, acc, false, false);
        }

        // Prefetch next timestep's xproj while WMMAs drain.
        float xpn[8] = {0.f, 0.f, 0.f, 0.f, 0.f, 0.f, 0.f, 0.f};
        if (t + 1 < T) {
#pragma unroll
            for (int r = 0; r < 8; ++r)
                xpn[r] = out[((long)(bBase + r + 8 * half) * T + (t + 1)) * H + ncol];
        }

        const int nxt = cur ^ 1;
#pragma unroll
        for (int r = 0; r < 8; ++r) {
            const int  m = r + 8 * half;           // local batch row
            const long b = bBase + m;
            const float v = tanhf(acc[r] + xp[r] + bias);
            out[(b * T + t) * H + ncol] = v;       // outs[b][t][ncol]
            hA[nxt][m][ncol] = (bf16_t)v;          // next step's A operand
            if (t == T - 1) hfin[b * H + ncol] = v;
        }
        __syncthreads();                            // hA[nxt] visible to all waves
        cur = nxt;
#pragma unroll
        for (int r = 0; r < 8; ++r) xp[r] = xpn[r];
    }
}

// ---------------------------------------------------------------------------
extern "C" void kernel_launch(void* const* d_in, const int* in_sizes, int n_in,
                              void* d_out, int out_size, void* d_ws, size_t ws_size,
                              hipStream_t stream) {
    const float* x   = (const float*)d_in[0];      // inputs       [B,T,I]
    const float* h0  = (const float*)d_in[1];      // hidden_state [B,H]
    const float* Wih = (const float*)d_in[2];      // W_ih         [H,I]
    const float* Whh = (const float*)d_in[3];      // W_hh         [H,H]
    const float* bih = (const float*)d_in[4];      // b_ih         [H]
    const float* bhh = (const float*)d_in[5];      // b_hh         [H]

    float* outs = (float*)d_out;                   // [B,T,H]
    float* hfin = outs + (size_t)B * T * H;        // [B,H]

    bf16_t* wsWih = (bf16_t*)d_ws;                 // 128 KB
    bf16_t* wsWhh = wsWih + (size_t)H * I;         // 128 KB

    cvt_weights_kernel<<<(H * I + 255) / 256, 256, 0, stream>>>(Wih, Whh, wsWih, wsWhh);
    xproj_kernel<<<(B * T) / 16, 256, 0, stream>>>(x, wsWih, bih, outs);
    scan_kernel<<<2, 512, 0, stream>>>(wsWhh, h0, bhh, outs, hfin);
}
